// Net_7060926234668
// MI455X (gfx1250) — matrix operand order repaired and checked by reading, not compiled, not run on hardware
//
#include <hip/hip_runtime.h>
#include <cmath>

// ---------------------------------------------------------------------------
// MI455X (gfx1250) implementation. wave32. All matmuls via
// v_wmma_f32_16x16x32_bf16 (bf16 inputs, f32 accumulate).
// Conv-chain activations stored bf16 (raw ushort bits) to halve HBM traffic.
// Conv weight tiles staged into LDS with the Tensor Data Mover when available.
// ---------------------------------------------------------------------------

#define DEVI __device__ __forceinline__

typedef __attribute__((ext_vector_type(16))) __bf16          v16bf;
typedef __attribute__((ext_vector_type(16))) unsigned short  v16u;
typedef __attribute__((ext_vector_type(8)))  float           v8f;
typedef __attribute__((ext_vector_type(4)))  unsigned int    v4u_;
typedef __attribute__((ext_vector_type(8)))  int             v8i_;
typedef __attribute__((ext_vector_type(4)))  int             v4i_;

#if defined(__has_builtin)
#if __has_builtin(__builtin_amdgcn_tensor_load_to_lds) && \
    __has_builtin(__builtin_amdgcn_s_wait_tensorcnt)
#define USE_TDM 1
#endif
#endif
#ifndef USE_TDM
#define USE_TDM 0
#endif

DEVI unsigned short f2bf(float f) {
  unsigned u = __builtin_bit_cast(unsigned, f);
  unsigned r = u + 0x7FFFu + ((u >> 16) & 1u);   // round-to-nearest-even
  return (unsigned short)(r >> 16);
}
DEVI float bf2f(unsigned short h) {
  unsigned u = ((unsigned)h) << 16;
  return __builtin_bit_cast(float, u);
}
DEVI v8f wmma_bf16(v16u a, v16u b, v8f c) {
  return __builtin_amdgcn_wmma_f32_16x16x32_bf16(
      false, __builtin_bit_cast(v16bf, a),
      false, __builtin_bit_cast(v16bf, b),
      (short)0, c, false, false);
}
DEVI float sigm(float x)   { return 1.f / (1.f + __expf(-x)); }
DEVI float logsig(float x) { return (x >= 0.f) ? -log1pf(__expf(-x)) : x - log1pf(__expf(x)); }
DEVI float gelu_t(float x) { float x3 = x*x*x; return 0.5f*x*(1.f + tanhf(0.79788456f*(x + 0.044715f*x3))); }

// ---------------------------------------------------------------------------
// utility kernels
// ---------------------------------------------------------------------------
__global__ void k_zero(float* p, long long n) {
  for (long long i = blockIdx.x * (long long)blockDim.x + threadIdx.x; i < n;
       i += (long long)gridDim.x * blockDim.x) p[i] = 0.f;
}

// f32 weight (R x C) -> bf16 padded [Kpad][Npad]; trans: src is (N,K)
__global__ void k_prep_w(const float* src, unsigned short* dst, int K, int N,
                         int Kpad, int Npad, int trans, float scale) {
  int tot = Kpad * Npad;
  for (int i = blockIdx.x * blockDim.x + threadIdx.x; i < tot;
       i += gridDim.x * blockDim.x) {
    int n = i % Npad, kk = i / Npad;
    float v = 0.f;
    if (kk < K && n < N)
      v = scale * (trans ? src[(long long)n * K + kk] : src[(long long)kk * N + n]);
    dst[i] = f2bf(v);
  }
}

// ---------------------------------------------------------------------------
// direct conv for Cin == 1 (f32 input -> bf16 out)
// ---------------------------------------------------------------------------
__global__ void k_conv_c1(const float* __restrict__ x, const float* __restrict__ w,
                          const float* __restrict__ bias, unsigned short* __restrict__ out,
                          int Lin, int Lout, int Cout, int K, int stride) {
  long long N = 512LL * Cout * Lout;
  for (long long i = blockIdx.x * (long long)blockDim.x + threadIdx.x; i < N;
       i += (long long)gridDim.x * blockDim.x) {
    int l = (int)(i % Lout);
    int o = (int)((i / Lout) % Cout);
    int b = (int)(i / ((long long)Lout * Cout));
    const float* xp = x + (long long)b * Lin + (long long)l * stride;
    const float* wp = w + (long long)o * K;
    float acc = bias[o];
    for (int t = 0; t < K; t++) acc += wp[t] * xp[t];
    out[i] = f2bf(acc);
  }
}

// ---------------------------------------------------------------------------
// implicit-GEMM conv via WMMA bf16.  M = output positions, N = out channels,
// K = Cin*k.  Weights (bf16 [Kpad][Opad]) staged in LDS (TDM if available).
// grid: (ceil(Lout/64), Opad/16, B), block 128 (4 waves, 1 row-tile each)
// ---------------------------------------------------------------------------
#define CONV_WAVES 4
__global__ void k_conv_wmma(const unsigned short* __restrict__ in,
                            const unsigned short* __restrict__ Wt,
                            const float* __restrict__ bias,
                            unsigned short* __restrict__ out,
                            int Lin, int Lout, int Cin, int Cout, int K,
                            int stride, int Kpad, int Opad) {
  __shared__ unsigned short sW[16384];
  int tid = threadIdx.x;
  int tot = Kpad * Opad;
#if USE_TDM
  if (tid < 32) {
    // Tensor DMA descriptor (ISA 8.3/8.4): 1 valid descriptor, 1-D tile of
    // `tot` 2-byte elements from global Wt into LDS sW.
    unsigned long long ga = (unsigned long long)(uintptr_t)Wt;
    unsigned lds = (unsigned)(uintptr_t)&sW[0];
    unsigned ne = (unsigned)tot;
    v4u_ g0;
    g0[0] = 1u;                                            // count=1, user D#
    g0[1] = lds;                                           // lds_addr
    g0[2] = (unsigned)(ga & 0xFFFFFFFFu);                  // global_addr[31:0]
    g0[3] = (unsigned)((ga >> 32) & 0x01FFFFFFu) | (2u << 30); // addr[56:32], type=2
    v8i_ g1;
    g1[0] = 0x00010000;                    // wg_mask=0, data_size=1 (2 bytes)
    g1[1] = (int)((ne & 0xFFFFu) << 16);   // tensor_dim0[15:0]
    g1[2] = (int)((ne >> 16) | (1u << 16));// tensor_dim0[31:16], tensor_dim1=1
    g1[3] = (int)((ne & 0xFFFFu) << 16);   // tile_dim0 = ne
    g1[4] = 0;                             // tile_dim1/2 unused (1-D)
    g1[5] = (int)ne;                       // tensor_dim0_stride[31:0]
    g1[6] = 0;
    g1[7] = 0;
    v4i_ gz4 = {0, 0, 0, 0};
    v8i_ gz8 = {0, 0, 0, 0, 0, 0, 0, 0};
    __builtin_amdgcn_tensor_load_to_lds(g0, g1, gz4, gz4, gz8, 0);
    __builtin_amdgcn_s_wait_tensorcnt(0);
  }
  __syncthreads();
#else
  for (int i = tid; i < tot; i += CONV_WAVES * 32) sW[i] = Wt[i];
  __syncthreads();
#endif

  int lane = tid & 31, wv = tid >> 5;
  int m0 = (blockIdx.x * CONV_WAVES + wv) * 16;
  int n0 = blockIdx.y * 16;
  int b  = blockIdx.z;
  if (m0 >= Lout) return;

  int row   = m0 + (lane & 15);
  bool rowv = row < Lout;
  int kbase = (lane < 16) ? 0 : 8;     // A-fragment K layout (ISA 7.12.2)
  int koff  = (lane < 16) ? 0 : 16;    // unused; B loads use the same kk as A
  int nn    = n0 + (lane & 15);
  const unsigned short* inb = in + (long long)b * Cin * Lin;
  int KT = Cin * K;
  if (rowv) __builtin_prefetch(&inb[(long long)row * stride], 0, 0);

  v8f acc = {};
  for (int ks = 0; ks < Kpad; ks += 32) {
    v16u a, bm;
#pragma unroll
    for (int h = 0; h < 16; h++) {
      int kk = ks + kbase + ((h >> 3) << 4) + (h & 7);
      unsigned short v = 0;
      if (rowv && kk < KT) {
        int ci = kk / K, tap = kk - ci * K;
        v = inb[(long long)ci * Lin + (long long)row * stride + tap];
      }
      a[h]  = v;
      bm[h] = sW[kk * Opad + nn];
    }
    acc = wmma_bf16(a, bm, acc);
  }
  if (nn < Cout) {
    float bs = bias ? bias[nn] : 0.f;
    int mb = (lane < 16) ? 0 : 8;
    unsigned short* ob = out + ((long long)b * Cout + nn) * Lout;
#pragma unroll
    for (int i = 0; i < 8; i++) {
      int m = m0 + mb + i;
      if (m < Lout) ob[m] = f2bf(acc[i] + bs);
    }
  }
}

// ---------------------------------------------------------------------------
// generic GEMM: Y[M,N] = epilogue( X[M,K](f32) @ Wt[Kpad,Npad](bf16) )
// Each wave: 16 rows x 32 cols (2 accumulators, A fragment reused).
// flags: 1 = +bias, 2 = +residual (ld=N), 4 = sigmoid
// grid: (M/64, Npad/32), block 128.  Npad must be a multiple of 32.
// ---------------------------------------------------------------------------
#define GW 4
DEVI void gemm_store(float* __restrict__ Y, const float* __restrict__ bias,
                     const float* __restrict__ res, v8f acc, int m0, int mb,
                     int nn, int M, int Nt, int flags) {
  if (nn >= Nt) return;
  float bs = (flags & 1) ? bias[nn] : 0.f;
#pragma unroll
  for (int i = 0; i < 8; i++) {
    int m = m0 + mb + i;
    if (m < M) {
      float v = acc[i] + bs;
      if (flags & 4) v = sigm(v);
      if (flags & 2) v += res[(long long)m * Nt + nn];
      Y[(long long)m * Nt + nn] = v;
    }
  }
}

__global__ void k_gemm(const float* __restrict__ X, const unsigned short* __restrict__ Wt,
                       const float* __restrict__ bias, const float* __restrict__ res,
                       float* __restrict__ Y, int M, int Kt, int Kpad, int Nt, int Npad,
                       int flags) {
  int lane = threadIdx.x & 31, wv = threadIdx.x >> 5;
  int m0 = (blockIdx.x * GW + wv) * 16;
  int n0 = blockIdx.y * 32;
  if (m0 >= M) return;
  int row = m0 + (lane & 15);
  bool rowv = row < M;
  int kbase = (lane < 16) ? 0 : 8;
  int koff  = (lane < 16) ? 0 : 16;
  int nn0 = n0 + (lane & 15);
  int nn1 = nn0 + 16;
  const float* xp = X + (long long)row * Kt;

  v8f acc0 = {}, acc1 = {};
  for (int ks = 0; ks < Kpad; ks += 32) {
    v16u a, bm0, bm1;
#pragma unroll
    for (int h = 0; h < 16; h++) {
      int kk = ks + kbase + ((h >> 3) << 4) + (h & 7);
      a[h] = (rowv && kk < Kt) ? f2bf(xp[kk]) : (unsigned short)0;
      const unsigned short* wr = Wt + (long long)kk * Npad;
      bm0[h] = wr[nn0];
      bm1[h] = wr[nn1];
    }
    acc0 = wmma_bf16(a, bm0, acc0);
    acc1 = wmma_bf16(a, bm1, acc1);
  }
  int mb = (lane < 16) ? 0 : 8;
  gemm_store(Y, bias, res, acc0, m0, mb, nn0, M, Nt, flags);
  gemm_store(Y, bias, res, acc1, m0, mb, nn1, M, Nt, flags);
}

// ---------------------------------------------------------------------------
// BN statistics + ACON
// ---------------------------------------------------------------------------
__global__ void k_rowstats(const unsigned short* __restrict__ x, float* rsum,
                           float* rsumsq, int L) {
  int row = blockIdx.x;
  const unsigned short* xp = x + (long long)row * L;
  float s = 0.f, s2 = 0.f;
  for (int i = threadIdx.x; i < L; i += blockDim.x) {
    float v = bf2f(xp[i]); s += v; s2 += v * v;
  }
  __shared__ float as[128], as2[128];
  as[threadIdx.x] = s; as2[threadIdx.x] = s2; __syncthreads();
  for (int off = 64; off > 0; off >>= 1) {
    if (threadIdx.x < off) { as[threadIdx.x] += as[threadIdx.x+off]; as2[threadIdx.x] += as2[threadIdx.x+off]; }
    __syncthreads();
  }
  if (threadIdx.x == 0) { rsum[row] = as[0]; rsumsq[row] = as2[0]; }
}

__global__ void k_chanfin(const float* rsum, const float* rsumsq, const float* g,
                          const float* bb, float* bnsc, float* bnsh, int C, int L) {
  int c = threadIdx.x;
  if (c >= C) return;
  float s = 0.f, s2 = 0.f;
  for (int b = 0; b < 512; b++) { s += rsum[b * C + c]; s2 += rsumsq[b * C + c]; }
  float inv = 1.f / (512.f * (float)L);
  float mu = s * inv, var = s2 * inv - mu * mu;
  float sc = g[c] * rsqrtf(var + 1e-5f);
  bnsc[c] = sc; bnsh[c] = bb[c] - mu * sc;
}

__global__ void k_aconbeta(const float* rsum, const float* bnsc, const float* bnsh,
                           const float* f1w, const float* f1b, const float* f2w,
                           const float* f2b, float* beta, int C, int L) {
  const int HID = 16;
  int b = blockIdx.x, t = threadIdx.x;
  __shared__ float ym[64], hd[32];
  if (t < C) ym[t] = (rsum[b * C + t] / (float)L) * bnsc[t] + bnsh[t];
  __syncthreads();
  if (t < HID) {
    float a = f1b[t];
    for (int c = 0; c < C; c++) a += f1w[t * C + c] * ym[c];
    hd[t] = a;
  }
  __syncthreads();
  if (t < C) {
    float a = f2b[t];
    for (int h = 0; h < HID; h++) a += f2w[t * HID + h] * hd[h];
    beta[b * C + t] = sigm(a);
  }
}

__global__ void k_bnacon(unsigned short* x, const float* bnsc, const float* bnsh,
                         const float* beta, const float* p1, const float* p2,
                         int C, int L) {
  long long N = 512LL * C * L;
  for (long long i = blockIdx.x * (long long)blockDim.x + threadIdx.x; i < N;
       i += (long long)gridDim.x * blockDim.x) {
    long long rc = i / L;
    int c = (int)(rc % C), b = (int)(rc / C);
    float v = bf2f(x[i]) * bnsc[c] + bnsh[c];
    float dd = (p1[c] - p2[c]) * v;
    x[i] = f2bf(dd * sigm(beta[b * C + c] * dd) + p2[c] * v);
  }
}

__global__ void k_mp2(const unsigned short* in, unsigned short* out, int C,
                      int Lin, int Lout) {
  long long N = 512LL * C * Lout;
  for (long long i = blockIdx.x * (long long)blockDim.x + threadIdx.x; i < N;
       i += (long long)gridDim.x * blockDim.x) {
    int l = (int)(i % Lout);
    long long rc = i / Lout;
    const unsigned short* p = in + rc * Lin + 2 * l;
    out[i] = f2bf(fmaxf(bf2f(p[0]), bf2f(p[1])));
  }
}

__global__ void k_mulew(const unsigned short* a, const unsigned short* b,
                        unsigned short* o, long long n) {
  for (long long i = blockIdx.x * (long long)blockDim.x + threadIdx.x; i < n;
       i += (long long)gridDim.x * blockDim.x)
    o[i] = f2bf(bf2f(a[i]) * bf2f(b[i]));
}

// ---------------------------------------------------------------------------
// coordinate attention pieces
// ---------------------------------------------------------------------------
__global__ void k_rowmax(const unsigned short* x, float* xmax, int L) {
  int row = blockIdx.x;
  const unsigned short* xp = x + (long long)row * L;
  float mv = -3.4e38f;
  for (int i = threadIdx.x; i < L; i += blockDim.x) mv = fmaxf(mv, bf2f(xp[i]));
  __shared__ float sm[128];
  sm[threadIdx.x] = mv; __syncthreads();
  for (int off = 64; off > 0; off >>= 1) {
    if (threadIdx.x < off) sm[threadIdx.x] = fmaxf(sm[threadIdx.x], sm[threadIdx.x+off]);
    __syncthreads();
  }
  if (threadIdx.x == 0) xmax[row] = sm[0];
}

__global__ void k_caconv1(const unsigned short* enc, const float* xw, const float* w,
                          const float* bias, unsigned short* y, int Cin, int Cout, int L) {
  long long N = 512LL * Cout * (L + 1);
  for (long long i = blockIdx.x * (long long)blockDim.x + threadIdx.x; i < N;
       i += (long long)gridDim.x * blockDim.x) {
    int pos = (int)(i % (L + 1));
    long long r = i / (L + 1);
    int o = (int)(r % Cout), b = (int)(r / Cout);
    float acc = bias[o];
    for (int c = 0; c < Cin; c++) {
      float v = (pos < L) ? bf2f(enc[((long long)b * Cin + c) * L + pos]) : xw[b * Cin + c];
      acc += w[o * Cin + c] * v;
    }
    y[i] = f2bf(acc);
  }
}

__global__ void k_caaw(const unsigned short* y, const float* w, const float* bias,
                       float* aw) {
  int n = 512 * 30;
  for (int i = blockIdx.x * blockDim.x + threadIdx.x; i < n; i += gridDim.x * blockDim.x) {
    int o = i % 30, b = i / 30;
    float a = bias[o];
    for (int c = 0; c < 6; c++) a += w[o * 6 + c] * bf2f(y[((long long)b * 6 + c) * 125 + 124]);
    aw[i] = sigm(a);
  }
}

__global__ void k_seqbuild(const unsigned short* enc, const float* aw, float* seq) {
  long long N = 30LL * 512 * 124;
  for (long long i = blockIdx.x * (long long)blockDim.x + threadIdx.x; i < N;
       i += (long long)gridDim.x * blockDim.x) {
    int l = (int)(i % 124);
    long long r = i / 124;
    int b = (int)(r % 512), c = (int)(r / 512);
    seq[i] = bf2f(enc[((long long)b * 30 + c) * 124 + l]) * aw[b * 30 + c];
  }
}

// ---------------------------------------------------------------------------
// layernorm over last dim (124); one wave per row
// ---------------------------------------------------------------------------
__global__ void k_lnorm(const float* x, const float* g, const float* bp, float* y,
                        int R, int Dd) {
  int row = blockIdx.x * 4 + (threadIdx.x >> 5);
  int lane = threadIdx.x & 31;
  if (row >= R) return;
  const float* xp = x + (long long)row * Dd;
  float s = 0.f, s2 = 0.f;
  for (int i = lane; i < Dd; i += 32) { float v = xp[i]; s += v; s2 += v * v; }
  for (int m = 16; m > 0; m >>= 1) { s += __shfl_xor(s, m, 32); s2 += __shfl_xor(s2, m, 32); }
  float mu = s / Dd, var = s2 / Dd - mu * mu;
  float rs = rsqrtf(var + 1e-5f);
  float* yp = y + (long long)row * Dd;
  for (int i = lane; i < Dd; i += 32) yp[i] = (xp[i] - mu) * rs * g[i] + bp[i];
}

// ---------------------------------------------------------------------------
// sLSTM step: rec = h_prev @ R[head]^T via one WMMA (K=32) per 16x16 tile,
// then gate math. grid (32 batch-tiles, 2 heads), block 128.
// ---------------------------------------------------------------------------
__global__ void k_slstm_step(const float* __restrict__ gx_s,
                             const unsigned short* __restrict__ Rt,
                             float* h, float* c, float* n, float* m,
                             float* __restrict__ hs_s) {
  int head = blockIdx.y;
  int b0 = blockIdx.x * 16;
  int lane = threadIdx.x & 31, wv = threadIdx.x >> 5;
  __shared__ float srec[16 * 128];

  int kbase = (lane < 16) ? 0 : 8;
  int koff  = (lane < 16) ? 0 : 16;
  int brow  = b0 + (lane & 15);
  v16u a;
#pragma unroll
  for (int hh = 0; hh < 16; hh++) {
    int kk = kbase + ((hh >> 3) << 4) + (hh & 7);
    a[hh] = f2bf(h[(long long)brow * 64 + head * 32 + kk]);
  }
  const unsigned short* R = Rt + head * 32 * 128;
#pragma unroll
  for (int t = 0; t < 2; t++) {
    int n0 = wv * 32 + t * 16;
    v16u bm;
#pragma unroll
    for (int hh = 0; hh < 16; hh++) bm[hh] = R[(kbase + ((hh >> 3) << 4) + (hh & 7)) * 128 + n0 + (lane & 15)];
    v8f acc = {};
    acc = wmma_bf16(a, bm, acc);
    int mb = (lane < 16) ? 0 : 8;
#pragma unroll
    for (int i = 0; i < 8; i++) srec[(mb + i) * 128 + n0 + (lane & 15)] = acc[i];
  }
  __syncthreads();

  int r  = threadIdx.x >> 3;          // 0..15  (batch row in tile)
  int dg = (threadIdx.x & 7) * 4;     // 0..28
  int b  = b0 + r;
#pragma unroll
  for (int j = 0; j < 4; j++) {
    int d = dg + j;
    long long gb = (long long)b * 256;
    float gi = gx_s[gb +   0 + head * 32 + d] + srec[r * 128 +  0 + d];
    float gf = gx_s[gb +  64 + head * 32 + d] + srec[r * 128 + 32 + d];
    float gz = gx_s[gb + 128 + head * 32 + d] + srec[r * 128 + 64 + d];
    float go = gx_s[gb + 192 + head * 32 + d] + srec[r * 128 + 96 + d];
    long long si = (long long)b * 64 + head * 32 + d;
    float lf = logsig(gf);
    float mo = m[si];
    float mn = fmaxf(lf + mo, gi);
    float ip = __expf(gi - mn);
    float fp = __expf(lf + mo - mn);
    float c2 = fp * c[si] + ip * tanhf(gz);
    float n2 = fp * n[si] + ip;
    float h2 = sigm(go) * c2 / fmaxf(n2, 1e-6f);
    c[si] = c2; n[si] = n2; m[si] = mn; h[si] = h2; hs_s[si] = h2;
  }
}

// ---------------------------------------------------------------------------
// mLSTM step: per (b, head) 32x32 C update + matvec. grid 1024, block 32.
// ---------------------------------------------------------------------------
__global__ void k_mlstm_step(const float* __restrict__ q, const float* __restrict__ k,
                             const float* __restrict__ v, const float* __restrict__ ifg,
                             const float* __restrict__ og,
                             float* C, float* n, float* m, float* __restrict__ hs) {
  int blk = blockIdx.x;
  int head = blk & 1, b = blk >> 1;
  int d = threadIdx.x;
  __shared__ float sq[32], sk[32];
  long long base = (long long)b * 64 + head * 32;
  float qd = q[base + d], kd = k[base + d], vd = v[base + d];
  sq[d] = qd; sk[d] = kd;
  __syncthreads();

  float it = ifg[(long long)b * 4 + head];
  float ft = ifg[(long long)b * 4 + 2 + head];
  float mo = m[(long long)b * 2 + head];
  float lf = logsig(ft);
  float mn = fmaxf(lf + mo, it);
  float ip = __expf(it - mn), fp = __expf(lf + mo - mn);

  float* Crow = C + ((long long)blk * 32 + d) * 32;
  float hval = 0.f;
#pragma unroll
  for (int j = 0; j < 32; j++) {
    float cc = fp * Crow[j] + ip * vd * sk[j];
    Crow[j] = cc;
    hval += cc * sq[j];
  }
  float n2 = fp * n[base + d] + ip * kd;
  n[base + d] = n2;
  float nq = n2 * qd;
  for (int mm = 16; mm > 0; mm >>= 1) nq += __shfl_xor(nq, mm, 32);
  float denom = fmaxf(fabsf(nq), 1.f);
  hs[base + d] = og[base + d] * hval / denom;
  __syncthreads();
  if (d == 0) m[(long long)b * 2 + head] = mn;
}

__global__ void k_glu(const float* U1, const float* U2, float* U, long long n) {
  for (long long i = blockIdx.x * (long long)blockDim.x + threadIdx.x; i < n;
       i += (long long)gridDim.x * blockDim.x)
    U[i] = gelu_t(U1[i]) * U2[i];
}

__global__ void k_meanfc(const float* __restrict__ seq3, const float* fcw,
                         const float* fcb, float* out) {
  int b = blockIdx.x, t = threadIdx.x;
  __shared__ float feat[30];
  if (t < 30) {
    const float* p = seq3 + ((long long)t * 512 + b) * 124;
    float s = 0.f;
    for (int l = 0; l < 124; l++) s += p[l];
    feat[t] = s / 124.f;
  }
  __syncthreads();
  if (t < 10) {
    float a = fcb[t];
    for (int s2 = 0; s2 < 30; s2++) a += fcw[t * 30 + s2] * feat[s2];
    out[b * 10 + t] = a;
  }
}

// ---------------------------------------------------------------------------
// host orchestration
// ---------------------------------------------------------------------------
extern "C" void kernel_launch(void* const* d_in, const int* in_sizes, int n_in,
                              void* d_out, int out_size, void* d_ws, size_t ws_size,
                              hipStream_t stream) {
  (void)in_sizes; (void)n_in; (void)out_size; (void)ws_size;
  // Flattened input indices (setup_inputs insertion order, depth-first):
  // 0:x | p1_1:1-10 p1_2:11-20 p2_1:21-30 p2_2:31-40 p2_4:41-50 p2_5:51-60
  //   (convbn order: w,b,g,bb, fc1_w,fc1_b,fc2_w,fc2_b, p1,p2)
  // ca: 61:conv1_w 62:conv1_b 63:g 64:bb 65:f1w 66:f1b 67:f2w 68:f2b 69:p1 70:p2
  //     71:convw_w 72:convw_b
  // s:  73:ln_g 74:ln_b 75:Wx 76:bx 77:R 78:Wu 79:Wv 80:Wd
  // m:  81:ln_g 82:ln_b 83:Wq 84:Wk 85:Wv 86:Wif 87:bif 88:Wo 89:Wd
  // 90:fc_w 91:fc_b
  auto F = [&](int i) { return (const float*)d_in[i]; };
  const float* x = F(0);

  char* base = (char*)d_ws;
  size_t off = 0;
  auto allocB = [&](size_t bytes) -> void* {
    void* p = base + off;
    off += (bytes + 255) & ~(size_t)255;
    return p;
  };
  auto allocU = [&](size_t n) { return (unsigned short*)allocB(n * 2); };
  auto allocF = [&](size_t n) { return (float*)allocB(n * 4); };

  unsigned short* bufA = allocU(26086400);        // 512*50*1019
  unsigned short* bufB = allocU(20766720);        // 512*40*1014
  unsigned short* bufC = allocU(10383360);        // 512*40*507
  unsigned short* p1b  = allocU(512 * 30 * 124);
  unsigned short* p2b  = allocU(512 * 30 * 124);
  unsigned short* encb = allocU(512 * 30 * 124);
  unsigned short* cay  = allocU(512 * 6 * 125);
  unsigned short* WtP1B = allocU(512 * 32);
  unsigned short* WtP2B = allocU(320 * 48);
  unsigned short* WtP2C = allocU(256 * 32);
  unsigned short* WtP2D = allocU(192 * 32);
  unsigned short* WtWx  = allocU(128 * 256);
  unsigned short* Rt    = allocU(2 * 32 * 128);
  unsigned short* WtWu  = allocU(64 * 96);
  unsigned short* WtWvS = allocU(64 * 96);
  unsigned short* WtWdS = allocU(96 * 128);
  unsigned short* WtQ   = allocU(128 * 64);
  unsigned short* WtK   = allocU(128 * 64);
  unsigned short* WtVm  = allocU(128 * 64);
  unsigned short* WtIF  = allocU(128 * 32);
  unsigned short* WtO   = allocU(128 * 64);
  unsigned short* WtWdM = allocU(64 * 128);
  float* rsum   = allocF(512 * 50);
  float* rsumsq = allocF(512 * 50);
  float* bnsc   = allocF(64);
  float* bnsh   = allocF(64);
  float* betab  = allocF(512 * 50);
  float* xw     = allocF(512 * 30);
  float* aw     = allocF(512 * 30);
  float* seq    = allocF(1904640);
  float* xn     = allocF(1904640);
  float* seq2   = allocF(1904640);
  float* seq3   = allocF(1904640);
  float* gx     = allocF(3932160);
  float* hsb    = allocF(983040);
  float* U1     = allocF(15360 * 85);
  float* U2     = allocF(15360 * 85);
  float* sstate = allocF(4 * 512 * 64);            // h,c,n,m
  float* qbuf   = allocF(983040);
  float* kbuf   = allocF(983040);
  float* vbuf   = allocF(983040);
  float* ogbuf  = allocF(983040);
  float* ifgb   = allocF(15360 * 4);
  float* hs2    = allocF(983040);
  float* mstate = allocF(1048576 + 32768 + 1024);  // C,n,m

  auto gEW = [](long long n) { return (unsigned)((n + 255) / 256); };
  auto prep = [&](const float* src, unsigned short* dst, int K, int N, int Kpad,
                  int Npad, int trans, float scale) {
    int tot = Kpad * Npad;
    k_prep_w<<<(tot + 255) / 256, 256, 0, stream>>>(src, dst, K, N, Kpad, Npad, trans, scale);
  };
  auto post = [&](unsigned short* buf, int C, int L, int bi) {
    k_rowstats<<<512 * C, 128, 0, stream>>>(buf, rsum, rsumsq, L);
    k_chanfin<<<1, 64, 0, stream>>>(rsum, rsumsq, F(bi + 2), F(bi + 3), bnsc, bnsh, C, L);
    k_aconbeta<<<512, 64, 0, stream>>>(rsum, bnsc, bnsh, F(bi + 4), F(bi + 5),
                                       F(bi + 6), F(bi + 7), betab, C, L);
    k_bnacon<<<gEW(512LL * C * L), 256, 0, stream>>>(buf, bnsc, bnsh, betab,
                                                     F(bi + 8), F(bi + 9), C, L);
  };

  // ---------------- p2 branch ----------------
  k_conv_c1<<<gEW(512LL * 50 * 1019), 256, 0, stream>>>(x, F(21), F(22), bufA, 1024, 1019, 50, 6, 1);
  post(bufA, 50, 1019, 21);
  prep(F(31), WtP2B, 300, 40, 320, 48, 1, 1.f);
  k_conv_wmma<<<dim3(16, 3, 512), 128, 0, stream>>>(bufA, WtP2B, F(32), bufB,
                                                    1019, 1014, 50, 40, 6, 1, 320, 48);
  post(bufB, 40, 1014, 31);
  k_mp2<<<gEW(512LL * 40 * 507), 256, 0, stream>>>(bufB, bufC, 40, 1014, 507);
  prep(F(41), WtP2C, 240, 30, 256, 32, 1, 1.f);
  k_conv_wmma<<<dim3(8, 2, 512), 128, 0, stream>>>(bufC, WtP2C, F(42), bufA,
                                                   507, 502, 40, 30, 6, 1, 256, 32);
  post(bufA, 30, 502, 41);
  prep(F(51), WtP2D, 180, 30, 192, 32, 1, 1.f);
  k_conv_wmma<<<dim3(4, 2, 512), 128, 0, stream>>>(bufA, WtP2D, F(52), bufB,
                                                   502, 249, 30, 30, 6, 2, 192, 32);
  post(bufB, 30, 249, 51);
  k_mp2<<<gEW(512LL * 30 * 124), 256, 0, stream>>>(bufB, p2b, 30, 249, 124);

  // ---------------- p1 branch ----------------
  k_conv_c1<<<gEW(512LL * 50 * 504), 256, 0, stream>>>(x, F(1), F(2), bufA, 1024, 504, 50, 18, 2);
  post(bufA, 50, 504, 1);
  prep(F(11), WtP1B, 500, 30, 512, 32, 1, 1.f);
  k_conv_wmma<<<dim3(4, 2, 512), 128, 0, stream>>>(bufA, WtP1B, F(12), bufB,
                                                   504, 248, 50, 30, 10, 2, 512, 32);
  post(bufB, 30, 248, 11);
  k_mp2<<<gEW(512LL * 30 * 124), 256, 0, stream>>>(bufB, p1b, 30, 248, 124);

  // ---------------- enc + coord attention ----------------
  k_mulew<<<gEW(512LL * 30 * 124), 256, 0, stream>>>(p1b, p2b, encb, 512LL * 30 * 124);
  k_rowmax<<<512 * 30, 128, 0, stream>>>(encb, xw, 124);
  k_caconv1<<<gEW(512LL * 6 * 125), 256, 0, stream>>>(encb, xw, F(61), F(62), cay, 30, 6, 124);
  post(cay, 6, 125, 61);
  k_caaw<<<gEW(512LL * 30), 256, 0, stream>>>(cay, F(71), F(72), aw);
  k_seqbuild<<<gEW(30LL * 512 * 124), 256, 0, stream>>>(encb, aw, seq);

  // ---------------- sLSTM block ----------------
  k_lnorm<<<3840, 128, 0, stream>>>(seq, F(73), F(74), xn, 15360, 124);
  prep(F(75), WtWx, 124, 256, 128, 256, 1, 1.f);
  k_gemm<<<dim3(240, 8), 128, 0, stream>>>(xn, WtWx, F(76), nullptr, gx,
                                           15360, 124, 128, 256, 256, 1);
  prep(F(77),             Rt,             32, 128, 32, 128, 1, 1.f);  // head 0
  prep(F(77) + 128 * 32,  Rt + 32 * 128,  32, 128, 32, 128, 1, 1.f);  // head 1
  k_zero<<<256, 256, 0, stream>>>(sstate, 4LL * 512 * 64);
  float* hst = sstate;
  float* cst = sstate + 32768;
  float* nst = sstate + 65536;
  float* mst = sstate + 98304;
  for (int s = 0; s < 30; s++)
    k_slstm_step<<<dim3(32, 2), 128, 0, stream>>>(gx + (long long)s * 512 * 256, Rt,
                                                  hst, cst, nst, mst,
                                                  hsb + (long long)s * 512 * 64);
  prep(F(78), WtWu,  64, 85, 64, 96, 0, 1.f);
  prep(F(79), WtWvS, 64, 85, 64, 96, 0, 1.f);
  k_gemm<<<dim3(240, 3), 128, 0, stream>>>(hsb, WtWu,  nullptr, nullptr, U1, 15360, 64, 64, 85, 96, 0);
  k_gemm<<<dim3(240, 3), 128, 0, stream>>>(hsb, WtWvS, nullptr, nullptr, U2, 15360, 64, 64, 85, 96, 0);
  k_glu<<<gEW(15360LL * 85), 256, 0, stream>>>(U1, U2, U1, 15360LL * 85);
  prep(F(80), WtWdS, 85, 124, 96, 128, 0, 1.f);
  k_gemm<<<dim3(240, 4), 128, 0, stream>>>(U1, WtWdS, nullptr, seq, seq2,
                                           15360, 85, 96, 124, 128, 2);

  // ---------------- mLSTM block ----------------
  k_lnorm<<<3840, 128, 0, stream>>>(seq2, F(81), F(82), xn, 15360, 124);
  prep(F(83), WtQ,  124, 64, 128, 64, 1, 1.f);
  prep(F(84), WtK,  124, 64, 128, 64, 1, 0.1767766953f);  // fold 1/sqrt(DH)
  prep(F(85), WtVm, 124, 64, 128, 64, 1, 1.f);
  prep(F(86), WtIF, 124, 4,  128, 32, 1, 1.f);
  prep(F(88), WtO,  124, 64, 128, 64, 1, 1.f);
  k_gemm<<<dim3(240, 2), 128, 0, stream>>>(xn, WtQ,  nullptr, nullptr, qbuf, 15360, 124, 128, 64, 64, 0);
  k_gemm<<<dim3(240, 2), 128, 0, stream>>>(xn, WtK,  nullptr, nullptr, kbuf, 15360, 124, 128, 64, 64, 0);
  k_gemm<<<dim3(240, 2), 128, 0, stream>>>(xn, WtVm, nullptr, nullptr, vbuf, 15360, 124, 128, 64, 64, 0);
  k_gemm<<<dim3(240, 2), 128, 0, stream>>>(xn, WtO,  nullptr, nullptr, ogbuf, 15360, 124, 128, 64, 64, 4);
  k_gemm<<<dim3(240, 1), 128, 0, stream>>>(xn, WtIF, F(87), nullptr, ifgb, 15360, 124, 128, 4, 32, 1);
  k_zero<<<1024, 256, 0, stream>>>(mstate, 1048576LL + 32768 + 1024);
  float* mC  = mstate;
  float* mns = mstate + 1048576;
  float* mms = mstate + 1048576 + 32768;
  for (int s = 0; s < 30; s++)
    k_mlstm_step<<<1024, 32, 0, stream>>>(qbuf + (long long)s * 32768,
                                          kbuf + (long long)s * 32768,
                                          vbuf + (long long)s * 32768,
                                          ifgb + (long long)s * 2048,
                                          ogbuf + (long long)s * 32768,
                                          mC, mns, mms,
                                          hs2 + (long long)s * 32768);
  prep(F(89), WtWdM, 64, 124, 64, 128, 0, 1.f);
  k_gemm<<<dim3(240, 4), 128, 0, stream>>>(hs2, WtWdM, nullptr, seq2, seq3,
                                           15360, 64, 64, 124, 128, 2);

  // ---------------- head ----------------
  k_meanfc<<<512, 32, 0, stream>>>(seq3, F(90), F(91), (float*)d_out);
}
